// ModularConstraintLinear_25005299597540
// MI455X (gfx1250) — matrix operand order repaired and checked by reading, not compiled, not run on hardware
//
#include <hip/hip_runtime.h>

typedef __attribute__((ext_vector_type(16))) __bf16 v16bf;
typedef __attribute__((ext_vector_type(8)))  float  v8f;
typedef __attribute__((ext_vector_type(4)))  unsigned int v4u;
typedef __attribute__((ext_vector_type(8)))  int v8i;
typedef __attribute__((ext_vector_type(4)))  int v4i;

#define BATCH   8192
#define IN_DIM  4096
#define OUT_DIM 4096
#define KS      64      // K per LDS stage (2 WMMA k-steps)
#define RP      72      // padded LDS row pitch in bf16 (144B = 9x16B, bank-conflict-free)
#define TM      128     // workgroup M tile (batch rows)
#define TN      128     // workgroup N tile (output cols)
#define NSTAGES (IN_DIM / KS)

union Frag  { v16bf v; uint4 q[2]; };
union Pack4 { __bf16 b[4]; uint2 q; };

static __device__ __forceinline__ v8f wmma_bf16(v16bf a, v16bf b, v8f c) {
  return __builtin_amdgcn_wmma_f32_16x16x32_bf16(false, a, false, b, (short)0, c,
                                                 false, false);
}

// ---------------------------------------------------------------------------
// TDM: one 2D tile DMA (global row-major, stride_elems bf16/row) into LDS at
// lds_addr. Rows are 64 bf16 = 32 DWORDs; pad_interval=4 (32 DW) with
// pad_amount=3 (4 DW = 16B) inserts 16B after every row -> LDS pitch 72 bf16,
// which makes all ds_load_b128 fragment reads bank-conflict-free.
// D# bitfields per CDNA5 ISA; 6-arg builtin (clang-23).
// ---------------------------------------------------------------------------
static __device__ __forceinline__ void tdm_load_2d(unsigned lds_addr,
                                                   const __bf16* gptr,
                                                   unsigned tile_x, unsigned tile_y,
                                                   unsigned stride_elems) {
  unsigned long long ga = (unsigned long long)gptr;
  const unsigned td = 1u << 20;          // tensor_dim0/1 >> tile dims (in-bounds)
  v4u g0;
  g0[0] = 1u;                            // count=1, user descriptor
  g0[1] = lds_addr;                      // lds_addr [63:32]
  g0[2] = (unsigned)ga;                  // global_addr [95:64]
  g0[3] = (unsigned)((ga >> 32) & 0x1FFFFFFu) | (2u << 30);  // ga[56:32] | type=2
  v8i g1;
  g1[0] = (int)((1u << 16)               // data_size=1 (2 bytes)
              | (1u << 20)               // pad_enable
              | (4u << 22)               // pad_interval: 32 DWORDs (= one row)
              | (3u << 25));             // pad_amount: 4 DWORDs (16B)
  g1[1] = (int)((td & 0xFFFFu) << 16);   // tensor_dim0[15:0] at bit 48
  g1[2] = (int)((td >> 16) | ((td & 0xFFFFu) << 16)); // td0[31:16] | td1[15:0]
  g1[3] = (int)((td >> 16) | (tile_x << 16));         // td1[31:16] | tile_dim0
  g1[4] = (int)tile_y;                   // tile_dim1; tile_dim2=0
  g1[5] = (int)stride_elems;             // tensor_dim0_stride[31:0]
  g1[6] = 0;
  g1[7] = 0;
  v4i z4 = (v4i){0, 0, 0, 0};
  v8i z8 = (v8i){0, 0, 0, 0, 0, 0, 0, 0};
  __builtin_amdgcn_tensor_load_to_lds(g0, g1, z4, z4, z8, 0);
}

// ---------------------------------------------------------------------------
// Kernel 1: symmetrize W by 12x12 blocks, split fp32 -> bf16 hi/lo.
// ---------------------------------------------------------------------------
__global__ __launch_bounds__(256) void symm_split_kernel(const float* __restrict__ W,
                                                         __bf16* __restrict__ Whi,
                                                         __bf16* __restrict__ Wlo) {
  unsigned idx = blockIdx.x * 256u + threadIdx.x;
  unsigned o = idx >> 12;
  unsigned i = idx & 4095u;
  float v = W[idx];
  if (o < 4092u && i < 4092u) {
    unsigned om = o % 12u, im = i % 12u;
    float p = W[(o - om + im) * (unsigned)IN_DIM + (i - im + om)];
    v = 0.5f * (v + p);
  }
  __bf16 hi = (__bf16)v;
  Whi[idx] = hi;
  Wlo[idx] = (__bf16)(v - (float)hi);
}

// ---------------------------------------------------------------------------
// Kernel 2: split X fp32 -> bf16 hi/lo (4 elements/thread, bandwidth pass).
// ---------------------------------------------------------------------------
__global__ __launch_bounds__(256) void x_split_kernel(const float* __restrict__ X,
                                                      __bf16* __restrict__ Xhi,
                                                      __bf16* __restrict__ Xlo) {
  size_t base = ((size_t)blockIdx.x * 256u + threadIdx.x) * 4u;
  float4 f = *(const float4*)(X + base);
  float ff[4] = {f.x, f.y, f.z, f.w};
  Pack4 ph, pl;
  #pragma unroll
  for (int e = 0; e < 4; ++e) {
    __bf16 hi = (__bf16)ff[e];
    ph.b[e] = hi;
    pl.b[e] = (__bf16)(ff[e] - (float)hi);
  }
  *(uint2*)(Xhi + base) = ph.q;
  *(uint2*)(Xlo + base) = pl.q;
}

// ---------------------------------------------------------------------------
// Kernel 3: out[b,o] = sum_i x[b,i]*W_sym[o,i] + bias[o]
// TDM double-buffered LDS feed (padded rows); 3-term bf16 split WMMA.
// 8 waves as 4(M) x 2(N); each wave owns a 32x64 output tile (2x4 subtiles).
// ---------------------------------------------------------------------------
__global__ __launch_bounds__(256) void gemm_sym_kernel(const __bf16* __restrict__ Xhi,
                                                       const __bf16* __restrict__ Xlo,
                                                       const __bf16* __restrict__ Whi,
                                                       const __bf16* __restrict__ Wlo,
                                                       const float* __restrict__ bias,
                                                       float* __restrict__ Out) {
  __shared__ __align__(16) __bf16 sXh[2][TM * RP];   // 2 x 18 KB
  __shared__ __align__(16) __bf16 sXl[2][TM * RP];   // 2 x 18 KB
  __shared__ __align__(16) __bf16 sWh[2][TN * RP];   // 2 x 18 KB
  __shared__ __align__(16) __bf16 sWl[2][TN * RP];   // 2 x 18 KB (144 KB total)

  const int tid  = threadIdx.x;
  const int lane = tid & 31;
  const int wave = tid >> 5;
  const int m_wave = (wave & 3) * 32;   // 4 waves across M
  const int n_wave = (wave >> 2) * 64;  // 2 waves across N
  const int half = lane >> 4;
  const int lr   = lane & 15;

  const int row0 = blockIdx.y * TM;
  const int col0 = blockIdx.x * TN;

  const __bf16* xh0 = Xhi + (size_t)row0 * IN_DIM;
  const __bf16* xl0 = Xlo + (size_t)row0 * IN_DIM;
  const __bf16* wh0 = Whi + (size_t)col0 * IN_DIM;
  const __bf16* wl0 = Wlo + (size_t)col0 * IN_DIM;

  v8f acc[2][4];
  #pragma unroll
  for (int i = 0; i < 2; ++i)
    #pragma unroll
    for (int j = 0; j < 4; ++j) acc[i][j] = (v8f){};

  // Prologue: DMA stage 0 into buffer 0 (TDM ignores EXEC; wave 0 issues).
  if (wave == 0) {
    tdm_load_2d((unsigned)(size_t)&sXh[0][0], xh0, KS, TM, IN_DIM);
    tdm_load_2d((unsigned)(size_t)&sXl[0][0], xl0, KS, TM, IN_DIM);
    tdm_load_2d((unsigned)(size_t)&sWh[0][0], wh0, KS, TN, IN_DIM);
    tdm_load_2d((unsigned)(size_t)&sWl[0][0], wl0, KS, TN, IN_DIM);
  }

  for (int s = 0; s < NSTAGES; ++s) {
    const int buf = s & 1;
    __syncthreads();   // all waves done reading buf^1 (stage s-1): safe to refill
    if (wave == 0) {
      if (s + 1 < NSTAGES) {
        const size_t k0 = (size_t)(s + 1) * KS;
        tdm_load_2d((unsigned)(size_t)&sXh[buf ^ 1][0], xh0 + k0, KS, TM, IN_DIM);
        tdm_load_2d((unsigned)(size_t)&sXl[buf ^ 1][0], xl0 + k0, KS, TM, IN_DIM);
        tdm_load_2d((unsigned)(size_t)&sWh[buf ^ 1][0], wh0 + k0, KS, TN, IN_DIM);
        tdm_load_2d((unsigned)(size_t)&sWl[buf ^ 1][0], wl0 + k0, KS, TN, IN_DIM);
        __builtin_amdgcn_s_wait_tensorcnt(4);  // stage s done; s+1 in flight
      } else {
        __builtin_amdgcn_s_wait_tensorcnt(0);
      }
    }
    __syncthreads();   // stage s data visible to all 8 waves

    #pragma unroll
    for (int kk = 0; kk < KS; kk += 32) {
      // A (16x32): lane<16 -> row lr, K {0..7}U{16..23}; lane>=16 -> K {8..15}U{24..31}
      Frag ah[2], al[2];
      #pragma unroll
      for (int i = 0; i < 2; ++i) {
        int m  = m_wave + i * 16 + lr;
        int o0 = m * RP + kk + half * 8;
        ah[i].q[0] = *(const uint4*)&sXh[buf][o0];
        ah[i].q[1] = *(const uint4*)&sXh[buf][o0 + 16];
        al[i].q[0] = *(const uint4*)&sXl[buf][o0];
        al[i].q[1] = *(const uint4*)&sXl[buf][o0 + 16];
      }
      // B (32x16): lane<16 -> col lr, K {0..7}U{16..23}; lane>=16 -> K {8..15}U{24..31}
      Frag bh[4], bl[4];
      #pragma unroll
      for (int j = 0; j < 4; ++j) {
        int n  = n_wave + j * 16 + lr;
        int ob = n * RP + kk + half * 8;
        bh[j].q[0] = *(const uint4*)&sWh[buf][ob];
        bh[j].q[1] = *(const uint4*)&sWh[buf][ob + 16];
        bl[j].q[0] = *(const uint4*)&sWl[buf][ob];
        bl[j].q[1] = *(const uint4*)&sWl[buf][ob + 16];
      }
      #pragma unroll
      for (int i = 0; i < 2; ++i)
        #pragma unroll
        for (int j = 0; j < 4; ++j) {
          acc[i][j] = wmma_bf16(ah[i].v, bh[j].v, acc[i][j]);
          acc[i][j] = wmma_bf16(al[i].v, bh[j].v, acc[i][j]);
          acc[i][j] = wmma_bf16(ah[i].v, bl[j].v, acc[i][j]);
        }
    }
  }

  // Epilogue: C layout (lane<16: M=v, N=lane; lane>=16: M=8+v, N=lane-16)
  #pragma unroll
  for (int j = 0; j < 4; ++j) {
    int gcol = col0 + n_wave + j * 16 + lr;
    float bv = bias[gcol];
    #pragma unroll
    for (int i = 0; i < 2; ++i) {
      int grow = row0 + m_wave + i * 16 + half * 8;
      #pragma unroll
      for (int v = 0; v < 8; ++v)
        Out[(size_t)(grow + v) * OUT_DIM + gcol] = acc[i][j][v] + bv;
    }
  }
}

extern "C" void kernel_launch(void* const* d_in, const int* in_sizes, int n_in,
                              void* d_out, int out_size, void* d_ws, size_t ws_size,
                              hipStream_t stream) {
  (void)in_sizes; (void)n_in; (void)out_size; (void)ws_size;
  const float* x    = (const float*)d_in[0];
  const float* w    = (const float*)d_in[1];
  const float* bias = (const float*)d_in[2];
  float* out = (float*)d_out;

  // workspace: Whi | Wlo | Xhi | Xlo  (32+32+64+64 = 192 MiB)
  __bf16* whi = (__bf16*)d_ws;
  __bf16* wlo = whi + (size_t)OUT_DIM * IN_DIM;
  __bf16* xhi = wlo + (size_t)OUT_DIM * IN_DIM;
  __bf16* xlo = xhi + (size_t)BATCH * IN_DIM;

  symm_split_kernel<<<(OUT_DIM * IN_DIM) / 256, 256, 0, stream>>>(w, whi, wlo);
  x_split_kernel<<<((size_t)BATCH * IN_DIM) / (256 * 4), 256, 0, stream>>>(x, xhi, xlo);
  gemm_sym_kernel<<<dim3(OUT_DIM / TN, BATCH / TM), 256, 0, stream>>>(xhi, xlo, whi,
                                                                      wlo, bias, out);
}
